// NodeBlock_4398046511956
// MI455X (gfx1250) — matrix operand order repaired and checked by reading, not compiled, not run on hardware
//
#include <hip/hip_runtime.h>

namespace {

constexpr int   kE   = 1600000;   // edges
constexpr int   kN   = 100000;    // nodes
constexpr float kEps = 1e-5f;

typedef __attribute__((ext_vector_type(16))) __bf16 v16bf;
typedef __attribute__((ext_vector_type(8)))  float  v8f;

union FragU { v16bf v; uint4 q[2]; };

__device__ __forceinline__ unsigned short f2bf(float f) {
  unsigned u = __builtin_bit_cast(unsigned, f);
  u += 0x7fffu + ((u >> 16) & 1u);          // round-to-nearest-even
  return (unsigned short)(u >> 16);
}
__device__ __forceinline__ float bf2f(unsigned short h) {
  unsigned u = ((unsigned)h) << 16;
  return __builtin_bit_cast(float, u);
}

// Load this wave's A fragments (all NK K-steps) once; they are invariant
// across all column tiles. sA: row-major [M][LDA] bf16 tile.
template <int LDA, int NK>
__device__ __forceinline__ void load_a_frags(const unsigned short* __restrict__ sA,
                                             int mt, int lane, FragU* fa) {
  const int lm = lane & 15, lh = lane >> 4;
  const unsigned short* ap = sA + (mt * 16 + lm) * LDA + lh * 8;
#pragma unroll
  for (int kk = 0; kk < NK; ++kk) {
    fa[kk].q[0] = *(const uint4*)(ap + kk * 32);       // K = kb .. kb+7
    fa[kk].q[1] = *(const uint4*)(ap + kk * 32 + 16);  // K = kb+16 .. kb+23
  }
}

// One 16x16 output tile using pre-loaded A fragments.
// sBt: transposed weights [n*LDB + k] bf16.
template <int LDB, int NK>
__device__ __forceinline__ v8f wmma_bt(const FragU* fa,
                                       const unsigned short* __restrict__ sBt,
                                       int nt, int lane, v8f c) {
  const int lm = lane & 15, lh = lane >> 4;
  const unsigned short* bp = sBt + (nt * 16 + lm) * LDB + lh * 8;
#pragma unroll
  for (int kk = 0; kk < NK; ++kk) {
    FragU fb;
    fb.q[0] = *(const uint4*)(bp + kk * 32);       // K = kb2 .. kb2+7
    fb.q[1] = *(const uint4*)(bp + kk * 32 + 16);  // K = kb2+16 .. kb2+23
    c = __builtin_amdgcn_wmma_f32_16x16x32_bf16(false, fa[kk].v, false, fb.v,
                                                (short)0, c, false, false);
  }
  return c;
}

// Load row-major f32 weights W[K][N] into LDS transposed bf16: sBt[n*K + k]
template <int K, int N>
__device__ __forceinline__ void load_w_t(const float* __restrict__ W,
                                         unsigned short* __restrict__ sBt, int t) {
  for (int i = t; i < K * N / 4; i += 256) {
    float4 w = ((const float4*)W)[i];
    int k = (i * 4) / N;
    int n = (i * 4) % N;
    sBt[(n + 0) * K + k] = f2bf(w.x);
    sBt[(n + 1) * K + k] = f2bf(w.y);
    sBt[(n + 2) * K + k] = f2bf(w.z);
    sBt[(n + 3) * K + k] = f2bf(w.w);
  }
}

__device__ __forceinline__ void cvt4(unsigned short* d, float4 v, float s) {
  d[0] = f2bf(v.x * s); d[1] = f2bf(v.y * s);
  d[2] = f2bf(v.z * s); d[3] = f2bf(v.w * s);
}

// ---------------------------------------------------------------- kernels ---

__global__ void k_zero(float* __restrict__ p, unsigned long long n) {
  unsigned long long i =
      (unsigned long long)blockIdx.x * blockDim.x + threadIdx.x;
  if (i < n) p[i] = 0.f;
}

// Pass 1: h = [x[src] | ea] @ W1a + b1a ; accumulate per-feature sum/sumsq.
__global__ __launch_bounds__(256) void k_edge_stats(
    const float* __restrict__ x, const int* __restrict__ ei,
    const float* __restrict__ ea, const float* __restrict__ W1a,
    const float* __restrict__ b1a, float* __restrict__ gSum,
    float* __restrict__ gSsq) {
  __shared__ unsigned short sA[128 * 128];
  __shared__ unsigned short sBt[128 * 128];
  __shared__ float sBias[128], sSum[128], sSsq[128];

  const int t = threadIdx.x;
  const int eBase = blockIdx.x * 128;

  load_w_t<128, 128>(W1a, sBt, t);
  if (t < 128) { sBias[t] = b1a[t]; sSum[t] = 0.f; sSsq[t] = 0.f; }
  {
    const int r = t >> 1, part = t & 1;
    const int e = eBase + r;
    const float4* src = part ? (const float4*)(ea + (size_t)e * 64)
                             : (const float4*)(x + (size_t)ei[e] * 64);
    unsigned short* dst = &sA[r * 128 + part * 64];
#pragma unroll
    for (int j = 0; j < 16; ++j) cvt4(dst + j * 4, src[j], 1.0f);
  }
  __syncthreads();

  const int w = t >> 5, lane = t & 31, lm = lane & 15;
  FragU fa[4];
  load_a_frags<128, 4>(sA, w, lane, fa);
  for (int nt = 0; nt < 8; ++nt) {
    v8f c; const float bv = sBias[nt * 16 + lm];
#pragma unroll
    for (int i = 0; i < 8; ++i) c[i] = bv;
    c = wmma_bt<128, 4>(fa, sBt, nt, lane, c);
    float s = 0.f, q = 0.f;
#pragma unroll
    for (int i = 0; i < 8; ++i) { float v = c[i]; s += v; q += v * v; }
    atomicAdd(&sSum[nt * 16 + lm], s);
    atomicAdd(&sSsq[nt * 16 + lm], q);
  }
  __syncthreads();
  if (t < 128) { atomicAdd(&gSum[t], sSum[t]); atomicAdd(&gSsq[t], sSsq[t]); }
}

__global__ void k_bn_finalize(const float* __restrict__ gSum,
                              const float* __restrict__ gSsq,
                              const float* __restrict__ gamma,
                              const float* __restrict__ beta,
                              float* __restrict__ scale,
                              float* __restrict__ shift, float invCnt) {
  int t = threadIdx.x;  // 128 threads
  float m = gSum[t] * invCnt;
  float v = gSsq[t] * invCnt - m * m;
  float sc = gamma[t] * rsqrtf(v + kEps);
  scale[t] = sc;
  shift[t] = beta[t] - m * sc;
}

// Pass 2: recompute GEMM1, apply BN1+ReLU, GEMM2 (+b1b), scatter-add to nodes.
__global__ __launch_bounds__(256) void k_edge_msg(
    const float* __restrict__ x, const int* __restrict__ ei,
    const float* __restrict__ ea, const float* __restrict__ W1a,
    const float* __restrict__ b1a, const float* __restrict__ W1b,
    const float* __restrict__ b1b, const float* __restrict__ scale,
    const float* __restrict__ shift, float* __restrict__ summed,
    float* __restrict__ counts) {
  __shared__ unsigned short sA[128 * 128];
  __shared__ unsigned short sB1[128 * 128];
  __shared__ unsigned short sB2[128 * 128];
  __shared__ float sBias1[128], sBias2[128], sScale[128], sShift[128];
  __shared__ int sDst[128];

  const int t = threadIdx.x;
  const int eBase = blockIdx.x * 128;

  load_w_t<128, 128>(W1a, sB1, t);
  load_w_t<128, 128>(W1b, sB2, t);
  if (t < 128) {
    sBias1[t] = b1a[t]; sBias2[t] = b1b[t];
    sScale[t] = scale[t]; sShift[t] = shift[t];
    sDst[t] = ei[kE + eBase + t];
  }
  {
    const int r = t >> 1, part = t & 1;
    const int e = eBase + r;
    const float4* src = part ? (const float4*)(ea + (size_t)e * 64)
                             : (const float4*)(x + (size_t)ei[e] * 64);
    unsigned short* dst = &sA[r * 128 + part * 64];
#pragma unroll
    for (int j = 0; j < 16; ++j) cvt4(dst + j * 4, src[j], 1.0f);
  }
  __syncthreads();

  const int w = t >> 5, lane = t & 31, lm = lane & 15, lh = lane >> 4;

  // GEMM1 into registers (finish all K reads before rewriting sA rows)
  v8f acc[8];
  {
    FragU fa[4];
    load_a_frags<128, 4>(sA, w, lane, fa);
#pragma unroll
    for (int nt = 0; nt < 8; ++nt) {
      v8f c; const float bv = sBias1[nt * 16 + lm];
#pragma unroll
      for (int i = 0; i < 8; ++i) c[i] = bv;
      acc[nt] = wmma_bt<128, 4>(fa, sB1, nt, lane, c);
    }
  }
  // BN1 + ReLU -> bf16 back into this wave's own 16 rows (no cross-wave hazard)
#pragma unroll
  for (int nt = 0; nt < 8; ++nt) {
    const int col = nt * 16 + lm;
    const float sc = sScale[col], sh = sShift[col];
#pragma unroll
    for (int i = 0; i < 8; ++i) {
      const int row = w * 16 + lh * 8 + i;
      sA[row * 128 + col] = f2bf(fmaxf(acc[nt][i] * sc + sh, 0.f));
    }
  }
  // GEMM2 + scatter-add to destination nodes (LDS ops are in-order per wave,
  // and each wave reads only its own rewritten rows)
  {
    FragU fa[4];
    load_a_frags<128, 4>(sA, w, lane, fa);
    for (int nt = 0; nt < 8; ++nt) {
      v8f c; const float bv = sBias2[nt * 16 + lm];
#pragma unroll
      for (int i = 0; i < 8; ++i) c[i] = bv;
      c = wmma_bt<128, 4>(fa, sB2, nt, lane, c);
      const int col = nt * 16 + lm;
#pragma unroll
      for (int i = 0; i < 8; ++i) {
        const int row = w * 16 + lh * 8 + i;
        atomicAdd(&summed[(size_t)sDst[row] * 128 + col], c[i]);
      }
    }
  }
  if (t < 128) atomicAdd(&counts[sDst[t]], 1.0f);
}

// Node pass 1: z = [x | agg | u[batch]] @ W2a + b2a ; BN2 stats ; store z (bf16)
__global__ __launch_bounds__(256) void k_node_stats(
    const float* __restrict__ x, const int* __restrict__ batch,
    const float* __restrict__ u, const float* __restrict__ summed,
    const float* __restrict__ counts, const float* __restrict__ W2a,
    const float* __restrict__ b2a, unsigned short* __restrict__ z1,
    float* __restrict__ gSum, float* __restrict__ gSsq) {
  __shared__ unsigned short sA[128 * 256];
  __shared__ unsigned short sBt[128 * 256];
  __shared__ float sBias[128], sSum[128], sSsq[128];

  const int t = threadIdx.x;
  const int nBase = blockIdx.x * 128;

  load_w_t<256, 128>(W2a, sBt, t);
  if (t < 128) { sBias[t] = b2a[t]; sSum[t] = 0.f; sSsq[t] = 0.f; }
  {
    const int r = t >> 1, part = t & 1;
    const int n = nBase + r;
    unsigned short* dst = &sA[r * 256 + part * 128];
    if (n < kN) {
      const float inv = 1.0f / fmaxf(counts[n], 1.0f);
      if (part == 0) {
        const float4* xs = (const float4*)(x + (size_t)n * 64);
        const float4* as = (const float4*)(summed + (size_t)n * 128);
#pragma unroll
        for (int j = 0; j < 16; ++j) cvt4(dst + j * 4, xs[j], 1.0f);
#pragma unroll
        for (int j = 0; j < 16; ++j) cvt4(dst + 64 + j * 4, as[j], inv);
      } else {
        const float4* as = (const float4*)(summed + (size_t)n * 128 + 64);
        const float4* us = (const float4*)(u + (size_t)batch[n] * 64);
#pragma unroll
        for (int j = 0; j < 16; ++j) cvt4(dst + j * 4, as[j], inv);
#pragma unroll
        for (int j = 0; j < 16; ++j) cvt4(dst + 64 + j * 4, us[j], 1.0f);
      }
    } else {
      for (int j = 0; j < 128; ++j) dst[j] = 0;
    }
  }
  __syncthreads();

  const int w = t >> 5, lane = t & 31, lm = lane & 15, lh = lane >> 4;

  v8f acc[8];
  {
    FragU fa[8];
    load_a_frags<256, 8>(sA, w, lane, fa);
#pragma unroll
    for (int nt = 0; nt < 8; ++nt) {
      v8f c; const float bv = sBias[nt * 16 + lm];
#pragma unroll
      for (int i = 0; i < 8; ++i) c[i] = bv;
      c = wmma_bt<256, 8>(fa, sBt, nt, lane, c);
      acc[nt] = c;
      float s = 0.f, q = 0.f;
#pragma unroll
      for (int i = 0; i < 8; ++i) {
        const int n = nBase + w * 16 + lh * 8 + i;
        if (n < kN) { float v = c[i]; s += v; q += v * v; }
      }
      atomicAdd(&sSum[nt * 16 + lm], s);
      atomicAdd(&sSsq[nt * 16 + lm], q);
    }
  }
  __syncthreads();  // everyone done reading sA
  // stage z (bf16, flat 128x128) in sA for coalesced stores
#pragma unroll
  for (int nt = 0; nt < 8; ++nt) {
    const int col = nt * 16 + lm;
#pragma unroll
    for (int i = 0; i < 8; ++i) {
      const int row = w * 16 + lh * 8 + i;
      sA[row * 128 + col] = f2bf(acc[nt][i]);
    }
  }
  __syncthreads();
  {
    const uint4* s4 = (const uint4*)sA;
    uint4* d4 = (uint4*)(z1 + (size_t)nBase * 128);
    const int rowsLeft = kN - nBase;
    for (int i = t; i < 2048; i += 256) {  // 16 uint4 per 128-col bf16 row
      if ((i >> 4) < rowsLeft) d4[i] = s4[i];
    }
  }
  if (t < 128) { atomicAdd(&gSum[t], sSum[t]); atomicAdd(&gSsq[t], sSsq[t]); }
}

// Node pass 2: out = relu(bn2(z)) @ W2b + b2b
__global__ __launch_bounds__(256) void k_node_out(
    const unsigned short* __restrict__ z1, const float* __restrict__ W2b,
    const float* __restrict__ b2b, const float* __restrict__ scale,
    const float* __restrict__ shift, float* __restrict__ out) {
  __shared__ unsigned short sA[128 * 128];
  __shared__ unsigned short sBt[64 * 128];
  __shared__ float sBias[64], sScale[128], sShift[128];

  const int t = threadIdx.x;
  const int nBase = blockIdx.x * 128;

  load_w_t<128, 64>(W2b, sBt, t);
  if (t < 64) sBias[t] = b2b[t];
  if (t < 128) { sScale[t] = scale[t]; sShift[t] = shift[t]; }
  __syncthreads();
  {
    const int r = t >> 1, part = t & 1;
    const int n = nBase + r;
    unsigned short* dst = &sA[r * 128 + part * 64];
    if (n < kN) {
      const uint4* s4 = (const uint4*)(z1 + (size_t)n * 128 + part * 64);
#pragma unroll
      for (int j = 0; j < 8; ++j) {
        uint4 v = s4[j];
        const unsigned short* hp = (const unsigned short*)&v;
#pragma unroll
        for (int e2 = 0; e2 < 8; ++e2) {
          const int col = part * 64 + j * 8 + e2;
          float f = bf2f(hp[e2]) * sScale[col] + sShift[col];
          dst[j * 8 + e2] = f2bf(fmaxf(f, 0.f));
        }
      }
    } else {
#pragma unroll
      for (int j = 0; j < 64; ++j) dst[j] = 0;
    }
  }
  __syncthreads();

  const int w = t >> 5, lane = t & 31, lm = lane & 15, lh = lane >> 4;
  FragU fa[4];
  load_a_frags<128, 4>(sA, w, lane, fa);
  for (int nt = 0; nt < 4; ++nt) {
    v8f c; const float bv = sBias[nt * 16 + lm];
#pragma unroll
    for (int i = 0; i < 8; ++i) c[i] = bv;
    c = wmma_bt<128, 4>(fa, sBt, nt, lane, c);
    const int col = nt * 16 + lm;
#pragma unroll
    for (int i = 0; i < 8; ++i) {
      const int row = w * 16 + lh * 8 + i;
      const int n = nBase + row;
      if (n < kN) out[(size_t)n * 64 + col] = c[i];
    }
  }
}

}  // namespace

extern "C" void kernel_launch(void* const* d_in, const int* in_sizes, int n_in,
                              void* d_out, int out_size, void* d_ws,
                              size_t ws_size, hipStream_t stream) {
  (void)in_sizes; (void)n_in; (void)out_size; (void)ws_size;
  const float* x      = (const float*)d_in[0];
  const int*   ei     = (const int*)d_in[1];
  const float* ea     = (const float*)d_in[2];
  const float* u      = (const float*)d_in[3];
  const int*   batch  = (const int*)d_in[4];
  const float* W1a    = (const float*)d_in[5];
  const float* b1a    = (const float*)d_in[6];
  const float* gamma1 = (const float*)d_in[7];
  const float* beta1  = (const float*)d_in[8];
  const float* W1b    = (const float*)d_in[9];
  const float* b1b    = (const float*)d_in[10];
  const float* W2a    = (const float*)d_in[11];
  const float* b2a    = (const float*)d_in[12];
  const float* gamma2 = (const float*)d_in[13];
  const float* beta2  = (const float*)d_in[14];
  const float* W2b    = (const float*)d_in[15];
  const float* b2b    = (const float*)d_in[16];
  float* out = (float*)d_out;

  // workspace layout
  unsigned short* z1 = (unsigned short*)d_ws;                    // kN*128 bf16
  float* summed = (float*)((char*)d_ws + (size_t)kN * 128 * 2);  // kN*128 f32
  float* counts = summed + (size_t)kN * 128;                     // kN f32
  float* st = counts + kN;                                       // 1024 f32
  float *sum1 = st,       *ssq1 = st + 128, *scl1 = st + 256, *shf1 = st + 384;
  float *sum2 = st + 512, *ssq2 = st + 640, *scl2 = st + 768, *shf2 = st + 896;

  const unsigned long long nz = (unsigned long long)kN * 128 + kN + 1024;
  k_zero<<<(unsigned)((nz + 255) / 256), 256, 0, stream>>>(summed, nz);

  k_edge_stats<<<kE / 128, 256, 0, stream>>>(x, ei, ea, W1a, b1a, sum1, ssq1);
  k_bn_finalize<<<1, 128, 0, stream>>>(sum1, ssq1, gamma1, beta1, scl1, shf1,
                                       1.0f / (float)kE);
  k_edge_msg<<<kE / 128, 256, 0, stream>>>(x, ei, ea, W1a, b1a, W1b, b1b, scl1,
                                           shf1, summed, counts);
  k_node_stats<<<(kN + 127) / 128, 256, 0, stream>>>(
      x, batch, u, summed, counts, W2a, b2a, z1, sum2, ssq2);
  k_bn_finalize<<<1, 128, 0, stream>>>(sum2, ssq2, gamma2, beta2, scl2, shf2,
                                       1.0f / (float)kN);
  k_node_out<<<(kN + 127) / 128, 256, 0, stream>>>(z1, W2b, b2b, scl2, shf2,
                                                   out);
}
